// SeqLabelerUMLSDisGaz_25211458028199
// MI455X (gfx1250) — matrix operand order repaired and checked by reading, not compiled, not run on hardware
//
#include <hip/hip_runtime.h>
#include <hip/hip_bf16.h>
#include <math.h>

// Problem constants (match reference)
#define T_LEN   512
#define NMOD    5
#define NELEM   6       // 5 modalities + null slot
#define HDIM    128
#define DKV     64
#define DCC     32
#define GDIM    512     // 4*H
#define KROW    384     // NELEM*DKV, row stride of keys/vals

// Multi-WGP scan configuration
#define NBLK    8       // cooperating workgroups
#define JSL     16      // H columns owned per block (128/8)
#define QSL     40      // q outputs owned per block (320/8)
#define PERH    73728   // packed halves per block slice (incl. pad to 18*8192B)
#define DYNLDS  147456  // bytes of dynamic LDS per scan block

typedef __attribute__((ext_vector_type(16))) _Float16 v16h;
typedef __attribute__((ext_vector_type(8)))  _Float16 v8h;
typedef __attribute__((ext_vector_type(2)))  _Float16 v2h;
typedef __attribute__((ext_vector_type(8)))  float    v8f;

// ---------------------------------------------------------------------------
// f32 -> f16 elementwise (bert activations)
// ---------------------------------------------------------------------------
__global__ void cvt_f16(const float* __restrict__ in, _Float16* __restrict__ out, int n) {
  int i = blockIdx.x * blockDim.x + threadIdx.x;
  if (i < n) out[i] = (_Float16)in[i];
}

// transpose (768 x 64) f32 -> (64 x 768) f16 so WMMA B fragments are contiguous
__global__ void cvt_w_t(const float* __restrict__ in, _Float16* __restrict__ out, int K) {
  int i = blockIdx.x * blockDim.x + threadIdx.x;
  int total = K * 64;
  if (i < total) {
    int k = i >> 6;
    int n = i & 63;
    out[n * K + k] = (_Float16)in[k * 64 + n];
  }
}

// ---------------------------------------------------------------------------
// Pack per-block scan weight slices, f32 -> f16, pair-interleaved layout
// [row/2][col][2] so the scan reads conflict-free 4B LDS words.
// Slice layout per block b (halves):
//   [0,20480)      Wih  cols owned by b      (64 rows x 320 cols)
//   [20480,61440)  Whh  cols owned by b      (128 rows x 320 cols)
//   [61440,66560)  Wq   outputs owned by b   (128 rows x 40 cols)
//   [66560,69120)  Wqc  outputs owned by b   (128 rows x 20 cols)
//   [69120,71680)  Wkc  outputs owned by b   (128 rows x 20 cols)
//   [71680,73728)  zero pad (copy granularity)
// Owned gate column for (m,gt,jj): g = gt*128 + b*JSL + jj, flat col = m*64+gt*16+jj
// ---------------------------------------------------------------------------
__global__ void pack_scan(const float* __restrict__ Wih, const float* __restrict__ Whh,
                          const float* __restrict__ Wq,  const float* __restrict__ Wqc,
                          const float* __restrict__ Wkc, _Float16* __restrict__ packedW) {
  int i = blockIdx.x * blockDim.x + threadIdx.x;
  if (i >= NBLK * PERH) return;
  int b = i / PERH, r = i % PERH;
  float val = 0.f;
  if (r < 20480) {                       // Wih slice [32][320][2]
    int p = r / 640, r2 = r % 640, col = r2 >> 1, lo = r2 & 1;
    int v = p * 2 + lo;
    int m = col >> 6, gt = (col >> 4) & 3, jj = col & 15;
    int g = gt * 128 + b * JSL + jj;
    val = Wih[(m * DKV + v) * GDIM + g];
  } else if (r < 61440) {                // Whh slice [64][320][2]
    int r1 = r - 20480;
    int p = r1 / 640, r2 = r1 % 640, col = r2 >> 1, lo = r2 & 1;
    int hh = p * 2 + lo;
    int m = col >> 6, gt = (col >> 4) & 3, jj = col & 15;
    int g = gt * 128 + b * JSL + jj;
    val = Whh[(m * HDIM + hh) * GDIM + g];
  } else if (r < 66560) {                // Wq slice [64][40][2]
    int r1 = r - 61440;
    int p = r1 / 80, r2 = r1 % 80, o = r2 >> 1, lo = r2 & 1;
    int hh = p * 2 + lo;
    int qi = b * QSL + o;
    int m = qi >> 6, k = qi & 63;
    val = Wq[(m * HDIM + hh) * DKV + k];
  } else if (r < 69120) {                // Wqc slice [64][20][2]
    int r1 = r - 66560;
    int p = r1 / 40, r2 = r1 % 40, o = r2 >> 1, lo = r2 & 1;
    int hh = p * 2 + lo;
    int m = o >> 2, kk = b * 4 + (o & 3);
    val = Wqc[(m * HDIM + hh) * DCC + kk];
  } else if (r < 71680) {                // Wkc slice [64][20][2]
    int r1 = r - 69120;
    int p = r1 / 40, r2 = r1 % 40, o = r2 >> 1, lo = r2 & 1;
    int hh = p * 2 + lo;
    int m = o >> 2, kk = b * 4 + (o & 3);
    val = Wkc[(m * HDIM + hh) * DCC + kk];
  }                                      // else: zero pad
  packedW[i] = (_Float16)val;
}

// ---------------------------------------------------------------------------
// WMMA GEMM: C[512 x 64] = A[512 x 768] (f16) @ Bt^T, fp32 accumulate.
// One wave = one 16x16 C tile; A and B both hold k = 16*(lane/16) + s (contiguous order).
// ---------------------------------------------------------------------------
__global__ void proj_wmma(const _Float16* __restrict__ A,
                          const _Float16* __restrict__ Bt,
                          float* __restrict__ out) {
  const int lane = threadIdx.x & 31;
  const int wave = threadIdx.x >> 5;
  const int tile = blockIdx.x * 8 + wave;   // 0..127
  const int tm = tile >> 2;
  const int tn = tile & 3;
  const int lh = lane >> 4;
  const int lm = lane & 15;

  union { v8h p[2]; v16h v; } fa;
  v8f acc = {};

  const _Float16* arow = A  + (tm * 16 + lm) * 768;
  const _Float16* bcol = Bt + (tn * 16 + lm) * 768 + lh * 16;

  for (int kt = 0; kt < 24; ++kt) {
    const int k0 = kt * 32;
    fa.p[0] = *(const v8h*)(arow + k0 + lh * 16);
    fa.p[1] = *(const v8h*)(arow + k0 + lh * 16 + 8);
    v16h bfrag = *(const v16h*)(bcol + k0);
    acc = __builtin_amdgcn_wmma_f32_16x16x32_f16(
        false, fa.v, false, bfrag, (short)0, acc, false, false);
  }

#pragma unroll
  for (int i = 0; i < 8; ++i) {
    int row = tm * 16 + lh * 8 + i;
    int col = tn * 16 + lm;
    out[row * KROW + col] = acc[i];   // slot 0
  }
}

// ---------------------------------------------------------------------------
// Small-modality projections (K = 50/20/2/2) + zero the null slot.
// ---------------------------------------------------------------------------
__global__ void small_proj(const int* __restrict__ umls_idx,
                           const int* __restrict__ pos_idx,
                           const float* __restrict__ dis,
                           const float* __restrict__ udis,
                           const float* __restrict__ umls_table,
                           const float* __restrict__ pos_table,
                           const float* __restrict__ Wk1, const float* __restrict__ Wv1,
                           const float* __restrict__ Wk2, const float* __restrict__ Wv2,
                           const float* __restrict__ Wk3, const float* __restrict__ Wv3,
                           const float* __restrict__ Wk4, const float* __restrict__ Wv4,
                           float* __restrict__ keys, float* __restrict__ vals) {
  const int total1 = 4 * 2 * T_LEN * DKV;
  const int total  = total1 + 2 * T_LEN * DKV;
  int i = blockIdx.x * blockDim.x + threadIdx.x;
  if (i >= total) return;

  if (i < total1) {
    int n = i & 63;
    int r = i >> 6;
    int t = r & 511;
    r >>= 9;
    int kv  = r & 1;
    int mod = r >> 1;
    const float* x; int d; const float* W;
    if (mod == 0)      { x = umls_table + (long)umls_idx[t] * 50; d = 50; W = kv ? Wv1 : Wk1; }
    else if (mod == 1) { x = pos_table  + (long)pos_idx[t]  * 20; d = 20; W = kv ? Wv2 : Wk2; }
    else if (mod == 2) { x = dis  + t * 2; d = 2; W = kv ? Wv3 : Wk3; }
    else               { x = udis + t * 2; d = 2; W = kv ? Wv4 : Wk4; }
    float s = 0.f;
    for (int k = 0; k < d; ++k) s += x[k] * W[k * 64 + n];
    float* dst = kv ? vals : keys;
    dst[t * KROW + (mod + 1) * 64 + n] = s;
  } else {
    int j = i - total1;
    int n = j & 63;
    int t = (j >> 6) & 511;
    int kv = j >> 15;
    float* dst = kv ? vals : keys;
    dst[t * KROW + 5 * 64 + n] = 0.f;
  }
}

// ---------------------------------------------------------------------------
// Inter-WGP barrier: monotonic counter, no reset races. Zeroed by init kernel
// at the start of every kernel_launch (graph-replay safe).
// ---------------------------------------------------------------------------
__global__ void init_sync(unsigned* p) {
  if (threadIdx.x == 0 && blockIdx.x == 0) *p = 0u;
}

__device__ __forceinline__ void gsync(unsigned* cnt, unsigned target) {
  __syncthreads();
  if (threadIdx.x == 0) {
    __threadfence();  // release: flush this WGP's global writes
    __hip_atomic_fetch_add(cnt, 1u, __ATOMIC_RELEASE, __HIP_MEMORY_SCOPE_AGENT);
    while (__hip_atomic_load(cnt, __ATOMIC_ACQUIRE, __HIP_MEMORY_SCOPE_AGENT) < target)
      __builtin_amdgcn_s_sleep(1);
    __threadfence();  // acquire: invalidate WGP$ so loads see remote writes
  }
  __syncthreads();
}

// ---------------------------------------------------------------------------
// Distributed MI-RIM scan: 8 cooperating workgroups, one per WGP.
// LDS-resident f16 weight slices (filled once via async LDS DMA); per step
// only q/h1/qc/kc/vc vectors cross WGPs (3 global barriers per step).
// c state never leaves its owning block; h kept redundantly per block.
// ---------------------------------------------------------------------------
__global__ __launch_bounds__(512)
void scan_mwgp(const float* __restrict__ keys, const float* __restrict__ vals,
               const _Float16* __restrict__ packedW,
               const float* __restrict__ b_lstm,
               const float* __restrict__ Wvc,
               float* __restrict__ qbuf,  float* __restrict__ h1buf,
               float* __restrict__ qcbuf, float* __restrict__ kcbuf,
               float* __restrict__ vcbuf,
               unsigned* __restrict__ barrier_cnt,
               float* __restrict__ hidden) {
  const int tid = threadIdx.x;
  const int b = blockIdx.x;

  extern __shared__ __align__(16) char smem_dyn[];
  const _Float16* wih_l = (const _Float16*)(smem_dyn);             // [32][320][2]
  const _Float16* whh_l = (const _Float16*)(smem_dyn + 40960);     // [64][320][2]
  const _Float16* wq_l  = (const _Float16*)(smem_dyn + 122880);    // [64][40][2]
  const _Float16* wqc_l = (const _Float16*)(smem_dyn + 133120);    // [64][20][2]
  const _Float16* wkc_l = (const _Float16*)(smem_dyn + 138240);    // [64][20][2]

  __shared__ float h_l[NMOD * HDIM], h1_l[NMOD * HDIM];
  __shared__ float c_l[NMOD * JSL];
  __shared__ float q_l[NMOD * DKV];
  __shared__ float attn_l[NMOD * NELEM];
  __shared__ float mask_l[NMOD];
  __shared__ float inp_l[NMOD * DKV];
  __shared__ float gates_l[NMOD * 64];
  __shared__ float qc_l[NMOD * DCC], kc_l[NMOD * DCC], vc_l[NMOD * HDIM];
  __shared__ float cattn_l[NMOD * NMOD];

  // ---- one-time: async-DMA this block's packed weight slice into LDS
  {
    const char* gsrc = (const char*)packedW + (size_t)b * DYNLDS;
    unsigned ldsBase = (unsigned)(size_t)(void*)smem_dyn;  // LDS byte offset
    for (int it = 0; it < 18; ++it) {
      unsigned off = (unsigned)(it * 8192 + tid * 16);
      unsigned laddr = ldsBase + off;
      const void* gaddr = gsrc + off;
      asm volatile("global_load_async_to_lds_b128 %0, %1, off"
                   :: "v"(laddr), "v"(gaddr) : "memory");
    }
    asm volatile("s_wait_asynccnt 0" ::: "memory");
  }
  for (int i = tid; i < NMOD * HDIM; i += 512) { h_l[i] = 0.f; h1_l[i] = 0.f; }
  if (tid < NMOD * JSL) c_l[tid] = 0.f;
  __syncthreads();

  const float inv_sdk = 0.125f;
  const float inv_sdc = 0.17677669529663687f;
  unsigned epoch = 0;

  for (int t = 0; t < T_LEN; ++t) {
    const float* ktp = keys + t * KROW;
    const float* vtp = vals + t * KROW;

    // ---- Stage 1: owned q outputs (LDS weights), publish to qbuf
    if (tid < QSL) {
      int qi = b * QSL + tid;
      int m = qi >> 6;
      const float* hm = h_l + m * HDIM;
      float acc = 0.f;
      for (int p = 0; p < 64; ++p) {
        v2h w = *(const v2h*)(wq_l + p * 80 + tid * 2);
        acc += hm[2 * p] * (float)w[0] + hm[2 * p + 1] * (float)w[1];
      }
      qbuf[qi] = acc;
    }
    gsync(barrier_cnt, NBLK * (++epoch));

    // ---- Stage 2 (redundant): attn softmax, top-K mask, inp
    if (tid < NMOD * DKV) q_l[tid] = qbuf[tid];
    __syncthreads();
    if (tid < NMOD) {
      int m = tid;
      float lg[NELEM], mx = -1e30f;
      for (int e = 0; e < NELEM; ++e) {
        float s = 0.f;
        for (int k = 0; k < DKV; ++k) s += q_l[m * DKV + k] * ktp[e * 64 + k];
        s *= inv_sdk; lg[e] = s; mx = fmaxf(mx, s);
      }
      float den = 0.f;
      for (int e = 0; e < NELEM; ++e) { lg[e] = expf(lg[e] - mx); den += lg[e]; }
      float rr = 1.f / den;
      for (int e = 0; e < NELEM; ++e) attn_l[m * NELEM + e] = lg[e] * rr;
    }
    __syncthreads();
    if (tid < NMOD) {   // 3 modules with smallest null-attn (top_k tie: low idx)
      float nu = attn_l[tid * NELEM + 5];
      int cntm = 0;
      for (int m2 = 0; m2 < NMOD; ++m2) {
        float nu2 = attn_l[m2 * NELEM + 5];
        if (nu2 < nu || (nu2 == nu && m2 < tid)) cntm++;
      }
      mask_l[tid] = (cntm < 3) ? 1.f : 0.f;
    }
    if (tid < NMOD * DKV) {
      int m = tid >> 6, v = tid & 63;
      float s = 0.f;
      for (int e = 0; e < NELEM; ++e) s += attn_l[m * NELEM + e] * vtp[e * 64 + v];
      inp_l[tid] = s;
    }
    __syncthreads();

    // ---- Stage 3: owned gate columns (LDS weights) + LSTM for owned j-slice
    if (tid < NMOD * 64) {
      int i = tid;
      int m = i >> 6, gt = (i >> 4) & 3, jj = i & 15;
      int g = gt * 128 + b * JSL + jj;
      float acc = b_lstm[m * GDIM + g];
      const float* im = inp_l + m * DKV;
      for (int p = 0; p < 32; ++p) {
        v2h w = *(const v2h*)(wih_l + p * 640 + i * 2);
        acc += im[2 * p] * (float)w[0] + im[2 * p + 1] * (float)w[1];
      }
      const float* hm = h_l + m * HDIM;
      for (int p = 0; p < 64; ++p) {
        v2h w = *(const v2h*)(whh_l + p * 640 + i * 2);
        acc += hm[2 * p] * (float)w[0] + hm[2 * p + 1] * (float)w[1];
      }
      gates_l[i] = acc;
    }
    __syncthreads();
    if (tid < NMOD * JSL) {   // 80 owned LSTM cells
      int m = tid >> 4, jj = tid & 15;
      int base = m * 64 + jj;
      float ig = 1.f / (1.f + expf(-gates_l[base]));
      float fg = 1.f / (1.f + expf(-gates_l[base + 16]));
      float gg = tanhf(gates_l[base + 32]);
      float og = 1.f / (1.f + expf(-gates_l[base + 48]));
      float cn = fg * c_l[tid] + ig * gg;
      float hn = og * tanhf(cn);
      float mk = mask_l[m];
      int hidx = m * HDIM + b * JSL + jj;
      float h1v = mk * hn + (1.f - mk) * h_l[hidx];
      c_l[tid] = mk * cn + (1.f - mk) * c_l[tid];
      h1buf[hidx] = h1v;
    }
    gsync(barrier_cnt, NBLK * (++epoch));

    // ---- Stage 4: gather full h1; owned qc/kc/vc outputs
    for (int i = tid; i < NMOD * HDIM; i += 512) h1_l[i] = h1buf[i];
    __syncthreads();
    if (tid < 20) {
      int m = tid >> 2, kk = b * 4 + (tid & 3);
      const float* hm = h1_l + m * HDIM;
      float acc = 0.f;
      for (int p = 0; p < 64; ++p) {
        v2h w = *(const v2h*)(wqc_l + p * 40 + tid * 2);
        acc += hm[2 * p] * (float)w[0] + hm[2 * p + 1] * (float)w[1];
      }
      qcbuf[m * DCC + kk] = acc;
    } else if (tid < 40) {
      int o = tid - 20;
      int m = o >> 2, kk = b * 4 + (o & 3);
      const float* hm = h1_l + m * HDIM;
      float acc = 0.f;
      for (int p = 0; p < 64; ++p) {
        v2h w = *(const v2h*)(wkc_l + p * 40 + o * 2);
        acc += hm[2 * p] * (float)w[0] + hm[2 * p + 1] * (float)w[1];
      }
      kcbuf[m * DCC + kk] = acc;
    } else if (tid >= 64 && tid < 64 + NMOD * JSL) {
      int o = tid - 64;
      int m = o >> 4, v = b * JSL + (o & 15);
      const float* hm = h1_l + m * HDIM;
      const float* wv = Wvc + (m * HDIM) * HDIM + v;   // f32, streamed from L2
      float acc = 0.f;
      for (int hh = 0; hh < HDIM; ++hh) acc += hm[hh] * wv[hh * HDIM];
      vcbuf[m * HDIM + v] = acc;
    }
    gsync(barrier_cnt, NBLK * (++epoch));

    // ---- Stage 5 (redundant): cattn softmax + h2; commit h; emit hidden[t]
    for (int i = tid; i < 2 * NMOD * DCC + NMOD * HDIM; i += 512) {
      if (i < NMOD * DCC)          qc_l[i] = qcbuf[i];
      else if (i < 2 * NMOD * DCC) kc_l[i - NMOD * DCC] = kcbuf[i - NMOD * DCC];
      else                         vc_l[i - 2 * NMOD * DCC] = vcbuf[i - 2 * NMOD * DCC];
    }
    __syncthreads();
    if (tid < NMOD) {
      int m = tid;
      float lg[NMOD], mx = -1e30f;
      for (int n = 0; n < NMOD; ++n) {
        float s = 0.f;
        for (int k = 0; k < DCC; ++k) s += qc_l[m * DCC + k] * kc_l[n * DCC + k];
        s *= inv_sdc; lg[n] = s; mx = fmaxf(mx, s);
      }
      float den = 0.f;
      for (int n = 0; n < NMOD; ++n) { lg[n] = expf(lg[n] - mx); den += lg[n]; }
      float rr = 1.f / den;
      for (int n = 0; n < NMOD; ++n) cattn_l[m * NMOD + n] = lg[n] * rr;
    }
    __syncthreads();
    for (int i = tid; i < NMOD * HDIM; i += 512) {
      int m = i >> 7, j = i & 127;
      float s = 0.f;
      for (int n = 0; n < NMOD; ++n) s += cattn_l[m * NMOD + n] * vc_l[n * HDIM + j];
      float h2 = h1_l[i] + mask_l[m] * s;
      h_l[i] = h2;
      if (b == 0) hidden[t * (NMOD * HDIM) + i] = h2;
    }
    __syncthreads();
  }
}

// ---------------------------------------------------------------------------
// Classifier: out[t,c] = hidden[t,:] @ Wcls[:,c] + bcls[c]
// ---------------------------------------------------------------------------
__global__ void cls_kernel(const float* __restrict__ hidden,
                           const float* __restrict__ Wcls,
                           const float* __restrict__ bcls,
                           float* __restrict__ out) {
  int i = blockIdx.x * blockDim.x + threadIdx.x;
  if (i >= T_LEN * 2) return;
  int t = i >> 1, cc = i & 1;
  const float* hrow = hidden + t * (NMOD * HDIM);
  float s = bcls[cc];
  for (int j = 0; j < NMOD * HDIM; ++j) s += hrow[j] * Wcls[j * 2 + cc];
  out[i] = s;
}

// ---------------------------------------------------------------------------
extern "C" void kernel_launch(void* const* d_in, const int* in_sizes, int n_in,
                              void* d_out, int out_size, void* d_ws, size_t ws_size,
                              hipStream_t stream) {
  (void)in_sizes; (void)n_in; (void)out_size; (void)ws_size;

  const float* bert       = (const float*)d_in[0];
  const int*   umls_idx   = (const int*)  d_in[1];
  const int*   pos_idx    = (const int*)  d_in[2];
  const float* dis        = (const float*)d_in[3];
  const float* udis       = (const float*)d_in[4];
  const float* umls_table = (const float*)d_in[5];
  const float* pos_table  = (const float*)d_in[6];
  const float* Wk0 = (const float*)d_in[7],  *Wv0 = (const float*)d_in[8];
  const float* Wk1 = (const float*)d_in[9],  *Wv1 = (const float*)d_in[10];
  const float* Wk2 = (const float*)d_in[11], *Wv2 = (const float*)d_in[12];
  const float* Wk3 = (const float*)d_in[13], *Wv3 = (const float*)d_in[14];
  const float* Wk4 = (const float*)d_in[15], *Wv4 = (const float*)d_in[16];
  const float* Wq     = (const float*)d_in[17];
  const float* Wih    = (const float*)d_in[18];
  const float* Whh    = (const float*)d_in[19];
  const float* b_lstm = (const float*)d_in[20];
  const float* Wqc    = (const float*)d_in[21];
  const float* Wkc    = (const float*)d_in[22];
  const float* Wvc    = (const float*)d_in[23];
  const float* Wcls   = (const float*)d_in[24];
  const float* bcls   = (const float*)d_in[25];

  // Workspace carve-out (256B aligned). Total ~5.1 MB.
  char* ws = (char*)d_ws;
  size_t off = 0;
  auto carve = [&](size_t bytes) -> void* {
    void* p = ws + off;
    off = (off + bytes + 255) & ~(size_t)255;
    return p;
  };
  float*     keys    = (float*)    carve((size_t)T_LEN * KROW * 4);
  float*     vals    = (float*)    carve((size_t)T_LEN * KROW * 4);
  float*     hidden  = (float*)    carve((size_t)T_LEN * NMOD * HDIM * 4);
  _Float16*  bert_h  = (_Float16*) carve((size_t)T_LEN * 768 * 2);
  _Float16*  wk0t    = (_Float16*) carve((size_t)64 * 768 * 2);
  _Float16*  wv0t    = (_Float16*) carve((size_t)64 * 768 * 2);
  _Float16*  packedW = (_Float16*) carve((size_t)NBLK * PERH * 2);
  float*     qbuf    = (float*)    carve(NMOD * DKV * 4);
  float*     h1buf   = (float*)    carve(NMOD * HDIM * 4);
  float*     qcbuf   = (float*)    carve(NMOD * DCC * 4);
  float*     kcbuf   = (float*)    carve(NMOD * DCC * 4);
  float*     vcbuf   = (float*)    carve(NMOD * HDIM * 4);
  unsigned*  barrier = (unsigned*) carve(256);

  // Phase 0: conversions + per-block weight slice packing
  cvt_f16   <<<dim3((T_LEN * 768 + 255) / 256), dim3(256), 0, stream>>>(bert, bert_h, T_LEN * 768);
  cvt_w_t   <<<dim3((768 * 64 + 255) / 256),    dim3(256), 0, stream>>>(Wk0, wk0t, 768);
  cvt_w_t   <<<dim3((768 * 64 + 255) / 256),    dim3(256), 0, stream>>>(Wv0, wv0t, 768);
  pack_scan <<<dim3((NBLK * PERH + 255) / 256), dim3(256), 0, stream>>>(Wih, Whh, Wq, Wqc, Wkc, packedW);

  // Phase 1: BERT projections via v_wmma_f32_16x16x32_f16 (slot 0)
  proj_wmma <<<dim3(16), dim3(256), 0, stream>>>(bert_h, wk0t, keys);
  proj_wmma <<<dim3(16), dim3(256), 0, stream>>>(bert_h, wv0t, vals);

  // Phase 1b: small modalities (slots 1-4) + null slot (slot 5)
  small_proj <<<dim3(1280), dim3(256), 0, stream>>>(
      umls_idx, pos_idx, dis, udis, umls_table, pos_table,
      Wk1, Wv1, Wk2, Wv2, Wk3, Wv3, Wk4, Wv4, keys, vals);

  // Phase 2: distributed MI-RIM scan (8 WGPs, LDS-resident weights,
  // async LDS fill, 3 inter-WGP barriers per step)
  init_sync <<<dim3(1), dim3(32), 0, stream>>>(barrier);
  scan_mwgp <<<dim3(NBLK), dim3(512), DYNLDS, stream>>>(
      keys, vals, packedW, b_lstm, Wvc,
      qbuf, h1buf, qcbuf, kcbuf, vcbuf, barrier, hidden);

  // Phase 3: classifier
  cls_kernel <<<dim3(4), dim3(256), 0, stream>>>(hidden, Wcls, bcls, (float*)d_out);
}
